// MockLlamaAttention_15659450761884
// MI455X (gfx1250) — matrix operand order repaired and checked by reading, not compiled, not run on hardware
//
#include <hip/hip_runtime.h>
#include <cstdint>
#include <cstddef>

// ---------------------------------------------------------------------------
// MockLlamaAttention on MI455X (gfx1250): bf16 WMMA pipeline + TDM staging
//   1) Q/K/V projections  : f32 GEMM via v_wmma_f32_16x16x32_bf16 (f32->bf16 in LDS)
//   2) RoPE + relayout    : rope(Q,K), scale folded into Q, bf16 [head][s][d]
//   3) Flash attention    : Q/K tiles staged by TENSOR_LOAD_TO_LDS (TDM, padded),
//                           QK^T and PV on WMMA, online softmax, causal mask
//   4) Output projection  : bf16 A (TDM-staged) x f32 W -> f32 out via WMMA
// ---------------------------------------------------------------------------

#define S_LEN 2048
#define HID   4096
#define NH    32
#define NKV   8
#define HD    128
#define KVH   (NKV * HD)   // 1024

#if defined(__has_builtin)
#if __has_builtin(__builtin_amdgcn_tensor_load_to_lds)
#define USE_TDM 1
#endif
#endif
#ifndef USE_TDM
#define USE_TDM 0
#endif

typedef __attribute__((ext_vector_type(16))) __bf16          v16bf;
typedef __attribute__((ext_vector_type(8)))  float           v8f;
typedef __attribute__((ext_vector_type(8)))  unsigned short  u16x8;
typedef __attribute__((ext_vector_type(4)))  unsigned short  u16x4;
typedef __attribute__((ext_vector_type(4)))  unsigned int    u32x4;
typedef __attribute__((ext_vector_type(8)))  int             i32x8;
typedef __attribute__((ext_vector_type(4)))  int             i32x4;

union FragBF {
  v16bf v;
  struct { u16x8 lo; u16x8 hi; } p;
};

__device__ __forceinline__ unsigned short f2b(float f) {
  unsigned int u = __float_as_uint(f);
  u += 0x7FFFu + ((u >> 16) & 1u);            // round-to-nearest-even
  return (unsigned short)(u >> 16);
}

// A-operand fragment (16-bit A 16x32 layout, ISA 7.12.2):
//   lane<16 : elems 0..7 -> K+0..7,  elems 8..15 -> K+16..23   (aoff = 0)
//   lane>=16: elems 0..7 -> K+8..15, elems 8..15 -> K+24..31   (aoff = 8)
__device__ __forceinline__ v16bf load_frag_a(const unsigned short* base) {
  FragBF f;
  f.p.lo = *reinterpret_cast<const u16x8*>(base);
  f.p.hi = *reinterpret_cast<const u16x8*>(base + 16);
  return f.v;
}

// B-operand fragment (same K order as A): lanes 0-15 hold K=0..7,16..23,
// lanes 16-31 hold K=8..15,24..31 (boff = 0 or 8).
__device__ __forceinline__ v16bf load_frag_b(const unsigned short* base) {
  FragBF f;
  f.p.lo = *reinterpret_cast<const u16x8*>(base);
  f.p.hi = *reinterpret_cast<const u16x8*>(base + 16);
  return f.v;
}

__device__ __forceinline__ v8f wmma_bf16(v16bf a, v16bf b, v8f c) {
  return __builtin_amdgcn_wmma_f32_16x16x32_bf16(
      /*neg_a=*/false, a, /*neg_b=*/false, b,
      /*c_mod=*/(short)0, c, /*reuse_a=*/false, /*reuse_b=*/false);
}

#if USE_TDM
// TENSOR_LOAD_TO_LDS of a 2-D bf16 tile (tile_d0 elems/row x tile_d1 rows)
// from a row-major tensor (row length tensor_d0 == stride0 elements), with
// hardware LDS row padding: after 2^(pad_icode+1) DWORDs stored, skip
// (pad_acode+1) DWORDs. D# layout per CDNA5 ISA 8.3/8.4 (groups 2/3 zero).
// This toolchain exposes the 6-arg builtin:
//   (uint32x4 g0, int32x8 g1, int32x4 g2, int32x4 g3, int32x8 extra, i32 cpol)
__device__ __forceinline__ void tdm_load_2d_bf16(
    unsigned lds_addr, unsigned long long gaddr,
    unsigned tensor_d0, unsigned tensor_d1,
    unsigned tile_d0, unsigned tile_d1, unsigned stride0,
    unsigned pad_icode, unsigned pad_acode) {
  u32x4 g0;
  g0[0] = 1u;                                         // count=1, user mode
  g0[1] = lds_addr;                                   // LDS byte address
  g0[2] = (unsigned)gaddr;                            // global addr [31:0]
  g0[3] = (unsigned)(gaddr >> 32) | (2u << 30);       // addr[56:32] | type=2
  i32x8 g1;
  g1[0] = (int)((1u << 16) |                          // data_size = 2 bytes
                (1u << 20) |                          // pad_enable
                (pad_icode << 22) | (pad_acode << 25));
  g1[1] = (int)((tensor_d0 & 0xFFFFu) << 16);         // tensor_dim0[15:0]
  g1[2] = (int)((tensor_d0 >> 16) | ((tensor_d1 & 0xFFFFu) << 16));
  g1[3] = (int)((tensor_d1 >> 16) | ((tile_d0 & 0xFFFFu) << 16));
  g1[4] = (int)(tile_d1 & 0xFFFFu);                   // tile_dim1; tile_dim2=0
  g1[5] = (int)stride0;                               // dim0 stride [31:0]
  g1[6] = 0;                                          // stride hi, dim1 stride
  g1[7] = 0;
  i32x4 z4 = {0, 0, 0, 0};
  i32x8 z8 = {0, 0, 0, 0, 0, 0, 0, 0};
  __builtin_amdgcn_tensor_load_to_lds(g0, g1, z4, z4, z8, 0);
}

__device__ __forceinline__ unsigned lds_byte_addr(const void* p) {
  return (unsigned)(uintptr_t)p;                      // low 32b = LDS offset
}
#endif

// ---------------------------------------------------------------------------
// GEMM: C[M,N] (f32) = A[M,K] x W[N,K]^T, bf16 WMMA math.
// BM=128, BN=128, BK=64; 256 threads = 8 waves in 4x2, each wave 32x64.
// A_BF16 path stages the A tile with the Tensor Data Mover.
// ---------------------------------------------------------------------------
template <bool A_BF16>
__global__ __launch_bounds__(256) void gemm_kernel(
    const void* __restrict__ Av, const float* __restrict__ W,
    float* __restrict__ C, int M, int N, int K) {
  constexpr int LDSW = 72;                       // 64 data + 8 pad shorts
  __shared__ unsigned short As[128 * LDSW];
  __shared__ unsigned short Bs[128 * LDSW];

  const int tid  = threadIdx.x;
  const int lane = tid & 31;
  const int wave = tid >> 5;
  const int l15  = lane & 15;
  const int half8 = (lane & 16) ? 8 : 0;
  const int aoff  = (lane & 16) ? 8 : 0;
  const int boff  = (lane & 16) ? 8 : 0;
  const int bm = blockIdx.y * 128;
  const int bn = blockIdx.x * 128;
  const int wm = (wave >> 1) * 32;
  const int wn = (wave & 1) * 64;

  const float* Af          = reinterpret_cast<const float*>(Av);
  const unsigned short* Ab = reinterpret_cast<const unsigned short*>(Av);

  v8f acc[2][4];
#pragma unroll
  for (int mt = 0; mt < 2; ++mt)
#pragma unroll
    for (int nt = 0; nt < 4; ++nt) acc[mt][nt] = {};

  for (int k0 = 0; k0 < K; k0 += 64) {
    if constexpr (!A_BF16) {
#pragma unroll
      for (int i = 0; i < 8; ++i) {
        int f   = tid + i * 256;                 // 2048 float4 chunks
        int row = f >> 4;
        int c4  = (f & 15) * 4;
        const float* src = Af + (size_t)(bm + row) * K + k0 + c4;
        if (k0 + 64 < K) __builtin_prefetch(src + 64, 0, 1);
        float4 v = *reinterpret_cast<const float4*>(src);
        u16x4 hh = { f2b(v.x), f2b(v.y), f2b(v.z), f2b(v.w) };
        *reinterpret_cast<u16x4*>(&As[row * LDSW + c4]) = hh;
      }
    } else {
#if USE_TDM
      if (wave == 0) {
        // 128 rows x 64 bf16; LDS rows 128B data + 16B pad -> codes (4, 3)
        tdm_load_2d_bf16(lds_byte_addr(&As[0]),
                         (unsigned long long)(uintptr_t)(Ab + (size_t)bm * K + k0),
                         (unsigned)K, 128u, 64u, 128u, (unsigned)K, 4u, 3u);
      }
#else
#pragma unroll
      for (int i = 0; i < 4; ++i) {
        int f   = tid + i * 256;                 // 1024 chunks of 8 shorts
        int row = f >> 3;
        int c8  = (f & 7) * 8;
        const unsigned short* src = Ab + (size_t)(bm + row) * K + k0 + c8;
        *reinterpret_cast<u16x8*>(&As[row * LDSW + c8]) =
            *reinterpret_cast<const u16x8*>(src);
      }
#endif
    }
#pragma unroll
    for (int i = 0; i < 8; ++i) {
      int f   = tid + i * 256;
      int row = f >> 4;
      int c4  = (f & 15) * 4;
      const float* src = W + (size_t)(bn + row) * K + k0 + c4;
      if (k0 + 64 < K) __builtin_prefetch(src + 64, 0, 1);
      float4 v = *reinterpret_cast<const float4*>(src);
      u16x4 hh = { f2b(v.x), f2b(v.y), f2b(v.z), f2b(v.w) };
      *reinterpret_cast<u16x4*>(&Bs[row * LDSW + c4]) = hh;
    }
#if USE_TDM
    if constexpr (A_BF16) {
      if (wave == 0) __builtin_amdgcn_s_wait_tensorcnt(0);
    }
#endif
    __syncthreads();

#pragma unroll
    for (int ks = 0; ks < 2; ++ks) {
      v16bf a[2], b[4];
#pragma unroll
      for (int mt = 0; mt < 2; ++mt)
        a[mt] = load_frag_a(&As[(wm + mt * 16 + l15) * LDSW + ks * 32 + aoff]);
#pragma unroll
      for (int nt = 0; nt < 4; ++nt)
        b[nt] = load_frag_b(&Bs[(wn + nt * 16 + l15) * LDSW + ks * 32 + boff]);
#pragma unroll
      for (int mt = 0; mt < 2; ++mt)
#pragma unroll
        for (int nt = 0; nt < 4; ++nt)
          acc[mt][nt] = wmma_bf16(a[mt], b[nt], acc[mt][nt]);
    }
    __syncthreads();
  }

#pragma unroll
  for (int mt = 0; mt < 2; ++mt)
#pragma unroll
    for (int nt = 0; nt < 4; ++nt)
#pragma unroll
      for (int i = 0; i < 8; ++i) {
        int row = bm + wm + mt * 16 + i + half8;
        int col = bn + wn + nt * 16 + l15;
        C[(size_t)row * N + col] = acc[mt][nt][i];
      }
}

// ---------------------------------------------------------------------------
// RoPE + bf16 relayout. grid = (48, S): u<32 -> Q head u (scaled by 1/sqrt(D)),
// u in [32,40) -> K head u-32, u in [40,48) -> V head u-40 (convert only).
// ---------------------------------------------------------------------------
__global__ __launch_bounds__(128) void rope_convert_kernel(
    const float* __restrict__ qraw, const float* __restrict__ kraw,
    const float* __restrict__ vraw, unsigned short* __restrict__ Qb,
    unsigned short* __restrict__ Kb, unsigned short* __restrict__ Vb) {
  const int s = blockIdx.y;
  const int u = blockIdx.x;
  const int d = threadIdx.x;                      // 0..127
  const int j = d & 63;
  const float LN_THETA = 9.210340371976184f;      // ln(10000)
  float inv = __expf(-((float)(2 * j) * (1.0f / 128.0f)) * LN_THETA);
  float ang = (float)s * inv;
  float sn, cs;
  __sincosf(ang, &sn, &cs);

  if (u < NH) {
    const float* base = qraw + (size_t)s * HID + u * HD;
    float x  = base[d];
    float xr = (d < 64) ? -base[d + 64] : base[d - 64];
    float o  = (x * cs + xr * sn) * 0.08838834764831845f;  // 1/sqrt(128)
    Qb[((size_t)u * S_LEN + s) * HD + d] = f2b(o);
  } else if (u < NH + NKV) {
    int hk = u - NH;
    const float* base = kraw + (size_t)s * KVH + hk * HD;
    float x  = base[d];
    float xr = (d < 64) ? -base[d + 64] : base[d - 64];
    Kb[((size_t)hk * S_LEN + s) * HD + d] = f2b(x * cs + xr * sn);
  } else {
    int hk = u - NH - NKV;
    Vb[((size_t)hk * S_LEN + s) * HD + d] =
        f2b(vraw[(size_t)s * KVH + hk * HD + d]);
  }
}

// ---------------------------------------------------------------------------
// Flash attention: grid = (S/128, NH). Block = 256 threads (8 waves).
// Q (128x128) and K (64x128) tiles staged by the Tensor Data Mover with
// hardware LDS padding; V transposed via VGPRs (overlaps in-flight TDM).
// ---------------------------------------------------------------------------
__global__ __launch_bounds__(256) void attn_kernel(
    const unsigned short* __restrict__ Qb, const unsigned short* __restrict__ Kb,
    const unsigned short* __restrict__ Vb, unsigned short* __restrict__ Ob) {
  constexpr int QLD = 136;                       // 128 data + 8 pad shorts
  constexpr int PLD = 72;                        // 64 data + 8 pad shorts
  __shared__ unsigned short Qs[128 * QLD];       // Q tile  128 x 128
  __shared__ unsigned short Ks[64 * QLD];        // K tile   64 x 128
  __shared__ unsigned short Vt[128 * PLD];       // V^T tile 128 x 64
  __shared__ unsigned short Ps[8 * 16 * PLD];    // per-wave P tiles 16 x 64

  const int tid  = threadIdx.x;
  const int lane = tid & 31;
  const int wave = tid >> 5;
  const int l15  = lane & 15;
  const int half8 = (lane & 16) ? 8 : 0;
  const int aoff  = (lane & 16) ? 8 : 0;
  const int boff  = (lane & 16) ? 8 : 0;

  const int qb = blockIdx.x;                     // query block 0..15
  const int h  = blockIdx.y;                     // head 0..31
  const int hk = h >> 2;                         // GQA: 4 Q heads per KV head

  const unsigned short* Qg = Qb + ((size_t)h * S_LEN + (size_t)qb * 128) * HD;
  const unsigned short* Kg = Kb + (size_t)hk * S_LEN * HD;
  const unsigned short* Vg = Vb + (size_t)hk * S_LEN * HD;

  // stage Q tile (128 x 128 bf16); LDS rows 256B data + 16B pad -> codes (5,3)
#if USE_TDM
  if (wave == 0) {
    tdm_load_2d_bf16(lds_byte_addr(&Qs[0]),
                     (unsigned long long)(uintptr_t)Qg,
                     128u, 128u, 128u, 128u, 128u, 5u, 3u);
  }
#else
#pragma unroll
  for (int i = 0; i < 8; ++i) {
    int c   = tid + i * 256;
    int row = c >> 4;
    int d8  = (c & 15) * 8;
    *reinterpret_cast<u16x8*>(&Qs[row * QLD + d8]) =
        *reinterpret_cast<const u16x8*>(Qg + (size_t)row * HD + d8);
  }
#endif

  v8f accO[8];
#pragma unroll
  for (int i = 0; i < 8; ++i) accO[i] = {};
  float m_i[8], l_i[8];
#pragma unroll
  for (int i = 0; i < 8; ++i) { m_i[i] = -3.0e38f; l_i[i] = 0.f; }

  const int qr    = wave * 16;
  const int qrow0 = qb * 128 + qr;
  const int nkb   = 2 * qb + 2;                  // key blocks covering causal span
  unsigned short* Pw = &Ps[wave * 16 * PLD];

  for (int kb = 0; kb < nkb; ++kb) {
    __syncthreads();                             // all reads of Ks/Vt done
#if USE_TDM
    if (wave == 0) {
      tdm_load_2d_bf16(lds_byte_addr(&Ks[0]),
                       (unsigned long long)(uintptr_t)(Kg + (size_t)kb * 64 * HD),
                       128u, 64u, 128u, 64u, 128u, 5u, 3u);
    }
#endif
    // stage V^T (128 x 64) via VGPRs; overlaps the in-flight TDM
#pragma unroll
    for (int i = 0; i < 4; ++i) {
      int c   = tid + i * 256;
      int row = c >> 4;
      int d8  = (c & 15) * 8;
#if !USE_TDM
      *reinterpret_cast<u16x8*>(&Ks[row * QLD + d8]) =
          *reinterpret_cast<const u16x8*>(Kg + (size_t)(kb * 64 + row) * HD + d8);
#endif
      u16x8 w = *reinterpret_cast<const u16x8*>(Vg + (size_t)(kb * 64 + row) * HD + d8);
#pragma unroll
      for (int jj = 0; jj < 8; ++jj) Vt[(d8 + jj) * PLD + row] = w[jj];
    }
#if USE_TDM
    if (wave == 0) __builtin_amdgcn_s_wait_tensorcnt(0);   // Q (first iter) + K
#endif
    __syncthreads();

    // scores S = Q (16x128) x K^T (128x64)
    v8f accS[4];
#pragma unroll
    for (int nt = 0; nt < 4; ++nt) accS[nt] = {};
#pragma unroll
    for (int kk = 0; kk < 4; ++kk) {
      v16bf qa = load_frag_a(&Qs[(qr + l15) * QLD + kk * 32 + aoff]);
#pragma unroll
      for (int nt = 0; nt < 4; ++nt) {
        v16bf kf = load_frag_b(&Ks[(nt * 16 + l15) * QLD + kk * 32 + boff]);
        accS[nt] = wmma_bf16(qa, kf, accS[nt]);
      }
    }

    // causal mask (diagonal blocks only)
    if (kb * 64 + 63 > qrow0) {
#pragma unroll
      for (int nt = 0; nt < 4; ++nt) {
        int colg = kb * 64 + nt * 16 + l15;
#pragma unroll
        for (int i = 0; i < 8; ++i)
          if (colg > qrow0 + i + half8) accS[nt][i] = -3.0e38f;
      }
    }

    // online softmax: row stats live across lanes 0-15 / 16-31 halves
    float mnew[8];
#pragma unroll
    for (int i = 0; i < 8; ++i) {
      float m = m_i[i];
#pragma unroll
      for (int nt = 0; nt < 4; ++nt) m = fmaxf(m, accS[nt][i]);
#pragma unroll
      for (int off = 1; off < 16; off <<= 1)
        m = fmaxf(m, __shfl_xor(m, off, 32));
      mnew[i] = m;
    }
    float fac[8], rs[8];
#pragma unroll
    for (int i = 0; i < 8; ++i) {
      fac[i] = __expf(m_i[i] - mnew[i]);
      m_i[i] = mnew[i];
      rs[i]  = 0.f;
    }
#pragma unroll
    for (int nt = 0; nt < 4; ++nt) {
#pragma unroll
      for (int i = 0; i < 8; ++i) {
        float p = __expf(accS[nt][i] - mnew[i]);
        rs[i] += p;
        // restage P in A-operand-friendly row-major (wave-private region)
        Pw[(i + half8) * PLD + nt * 16 + l15] = f2b(p);
      }
    }
#pragma unroll
    for (int i = 0; i < 8; ++i) {
      float r = rs[i];
#pragma unroll
      for (int off = 1; off < 16; off <<= 1) r += __shfl_xor(r, off, 32);
      l_i[i] = l_i[i] * fac[i] + r;
#pragma unroll
      for (int dt = 0; dt < 8; ++dt) accO[dt][i] *= fac[i];
    }
    asm volatile("s_wait_dscnt 0x0" ::: "memory");  // P stores -> P loads (same wave)

    // O += P (16x64) x V (64x128)
#pragma unroll
    for (int kk = 0; kk < 2; ++kk) {
      v16bf pa = load_frag_a(&Pw[l15 * PLD + kk * 32 + aoff]);
#pragma unroll
      for (int dt = 0; dt < 8; ++dt) {
        v16bf vf = load_frag_b(&Vt[(dt * 16 + l15) * PLD + kk * 32 + boff]);
        accO[dt] = wmma_bf16(pa, vf, accO[dt]);
      }
    }
  }

  // finalize: O /= l, write bf16 [s][h*128+d]
  float invl[8];
#pragma unroll
  for (int i = 0; i < 8; ++i) invl[i] = 1.0f / l_i[i];
#pragma unroll
  for (int dt = 0; dt < 8; ++dt)
#pragma unroll
    for (int i = 0; i < 8; ++i) {
      float o  = accO[dt][i] * invl[i];
      int rowg = qrow0 + i + half8;
      int colg = h * HD + dt * 16 + l15;
      Ob[(size_t)rowg * HID + colg] = f2b(o);
    }
}

// ---------------------------------------------------------------------------
extern "C" void kernel_launch(void* const* d_in, const int* in_sizes, int n_in,
                              void* d_out, int out_size, void* d_ws, size_t ws_size,
                              hipStream_t stream) {
  (void)in_sizes; (void)n_in; (void)out_size; (void)ws_size;
  const float* x  = (const float*)d_in[0];
  const float* wq = (const float*)d_in[1];
  const float* wk = (const float*)d_in[2];
  const float* wv = (const float*)d_in[3];
  const float* wo = (const float*)d_in[4];
  float* out = (float*)d_out;

  char* ws = (char*)d_ws;
  float* qraw = (float*)ws;            ws += (size_t)S_LEN * HID * 4;
  float* kraw = (float*)ws;            ws += (size_t)S_LEN * KVH * 4;
  float* vraw = (float*)ws;            ws += (size_t)S_LEN * KVH * 4;
  unsigned short* Qb = (unsigned short*)ws; ws += (size_t)S_LEN * HID * 2;
  unsigned short* Kb = (unsigned short*)ws; ws += (size_t)S_LEN * KVH * 2;
  unsigned short* Vb = (unsigned short*)ws; ws += (size_t)S_LEN * KVH * 2;
  unsigned short* Ob = (unsigned short*)ws; ws += (size_t)S_LEN * HID * 2;

  const dim3 blk(256);
  // projections: C = X @ W^T
  gemm_kernel<false><<<dim3(HID / 128, S_LEN / 128), blk, 0, stream>>>(
      x, wq, qraw, S_LEN, HID, HID);
  gemm_kernel<false><<<dim3(KVH / 128, S_LEN / 128), blk, 0, stream>>>(
      x, wk, kraw, S_LEN, KVH, HID);
  gemm_kernel<false><<<dim3(KVH / 128, S_LEN / 128), blk, 0, stream>>>(
      x, wv, vraw, S_LEN, KVH, HID);
  // rope + bf16 relayout
  rope_convert_kernel<<<dim3(NH + 2 * NKV, S_LEN), dim3(128), 0, stream>>>(
      qraw, kraw, vraw, Qb, Kb, Vb);
  // flash attention
  attn_kernel<<<dim3(S_LEN / 128, NH), blk, 0, stream>>>(Qb, Kb, Vb, Ob);
  // output projection -> f32 result
  gemm_kernel<true><<<dim3(HID / 128, S_LEN / 128), blk, 0, stream>>>(
      Ob, wo, out, S_LEN, HID, HID);
}
